// ptTransformerBlock_58832462020793
// MI455X (gfx1250) — matrix operand order repaired and checked by reading, not compiled, not run on hardware
//
#include <hip/hip_runtime.h>
#include <hip/hip_bf16.h>
#include <math.h>
#include <stdint.h>

// ---------------------------------------------------------------------------
// Point Transformer block for MI455X (gfx1250), wave32 + WMMA bf16 + TDM.
// B=4, N=4096, K=16, D_POINTS=64, D_MODEL=128.
// ---------------------------------------------------------------------------

typedef __bf16 bf16;
typedef __bf16  v16bf __attribute__((ext_vector_type(16)));
typedef __bf16  v8bf  __attribute__((ext_vector_type(8)));
typedef float   v8f   __attribute__((ext_vector_type(8)));
typedef unsigned int v4u __attribute__((ext_vector_type(4)));
typedef int v8i __attribute__((ext_vector_type(8)));
typedef int v4i __attribute__((ext_vector_type(4)));

#define BN   4
#define NP   4096
#define KNN  16
#define DP   64
#define DM   128
#define SCALE 0.088388347648318447f   // 1/sqrt(128)

#if __has_builtin(__builtin_amdgcn_tensor_load_to_lds)
#define USE_TDM 1
#else
#define USE_TDM 0
#endif

// ---------------- fragment helpers (layouts per CDNA5 ISA 7.12.2) ----------

// A-matrix 16x32 bf16: lane m = lane&15 (both halves same row).
// hl=0: elems 0..7 -> K 0..7,  elems 8..15 -> K 16..23
// hl=1: elems 0..7 -> K 8..15, elems 8..15 -> K 24..31
__device__ inline v16bf load_afrag_lds(const bf16* base, int m, int kk, int hl) {
    v8bf lo = *(const v8bf*)(base + m * DM + kk * 32 + hl * 8);
    v8bf hi = *(const v8bf*)(base + m * DM + kk * 32 + 16 + hl * 8);
    v16bf a;
#pragma unroll
    for (int e = 0; e < 8; ++e) { a[e] = lo[e]; a[8 + e] = hi[e]; }
    return a;
}

// B-matrix 32x16 bf16 from transposed (col-major) weight WT[col][K]:
// lane col n = lane&15, elems 0..7 -> K 8*hl.., elems 8..15 -> K 16+8*hl...
__device__ inline v16bf load_bfrag_g(const bf16* WT, int col, int kk, int hl, int kdim) {
    v8bf lo = *(const v8bf*)(WT + col * kdim + kk * 32 + hl * 8);
    v8bf hi = *(const v8bf*)(WT + col * kdim + kk * 32 + 16 + hl * 8);
    v16bf b;
#pragma unroll
    for (int e = 0; e < 8; ++e) { b[e] = lo[e]; b[8 + e] = hi[e]; }
    return b;
}

__device__ inline v16bf load_bfrag_l(const bf16* WT, int col, int kk, int hl) {
    const bf16* p = WT + col * DM + kk * 32 + hl * 8;
    v8bf lo = *(const v8bf*)(p);
    v8bf hi = *(const v8bf*)(p + 16);
    v16bf b;
#pragma unroll
    for (int e = 0; e < 8; ++e) { b[e] = lo[e]; b[8 + e] = hi[e]; }
    return b;
}

// ---------------- TDM: 1-D global -> LDS copy (elements of 2 bytes) --------
// D# per CDNA5 ISA ch.8: group0 {count=1, lds_addr, global_addr, type=2},
// group1 {data_size=2B, tensor_dim0=n, tensor_dim1=1, tile_dim0=n,
//         tile_dim1=1, tensor_dim0_stride=n}.  Single descriptor, no cluster.
#if USE_TDM
__device__ inline void tdm_load_1d_bf16(const bf16* gsrc, unsigned lds_off,
                                        unsigned nelems) {
    unsigned long long ga = (unsigned long long)(uintptr_t)gsrc;
    v4u g0;
    g0[0] = 1u;                                            // count=1 (valid)
    g0[1] = lds_off;                                       // lds_addr (bytes)
    g0[2] = (unsigned)(ga & 0xFFFFFFFFu);                  // global_addr lo
    g0[3] = (unsigned)((ga >> 32) & 0x01FFFFFFu) | (2u << 30); // hi | type=2
    v8i g1;
    g1[0] = (int)(1u << 16);                               // data_size=1 (2B)
    g1[1] = (int)((nelems & 0xFFFFu) << 16);               // tensor_dim0 lo16
    g1[2] = (int)(((nelems >> 16) & 0xFFFFu) | (1u << 16)); // td0 hi | td1=1 lo
    g1[3] = (int)((nelems & 0xFFFFu) << 16);               // td1 hi=0 | tile_dim0
    g1[4] = 1;                                             // tile_dim1=1
    g1[5] = (int)nelems;                                   // td0_stride lo32
    g1[6] = 0;
    g1[7] = 0;
    v4i z4; z4[0] = 0; z4[1] = 0; z4[2] = 0; z4[3] = 0;
    v8i z8;
#pragma unroll
    for (int e = 0; e < 8; ++e) z8[e] = 0;
    // 6-arg form (clang-23 / therock headers): groups 0..3 + extra + cpol
    __builtin_amdgcn_tensor_load_to_lds(g0, g1, z4, z4, z8, 0);
}
#endif

// ---------------- kernel 0: weight transpose + f32->bf16 -------------------
__global__ void transpose_bf16_kernel(const float* __restrict__ src,
                                      bf16* __restrict__ dst, int R, int C) {
    int o = blockIdx.x * blockDim.x + threadIdx.x;
    if (o >= R * C) return;
    int c = o / R, r = o - c * R;
    dst[o] = (bf16)src[r * C + c];
}

// ---------------- kernel 1: KNN (top-16 by squared distance) ---------------
__global__ void knn_kernel(const float* __restrict__ xyz, int* __restrict__ idx) {
    __shared__ float sx[256], sy[256], sz[256];
    int b = blockIdx.y;
    int i = blockIdx.x * 256 + threadIdx.x;
    const float* base = xyz + (size_t)b * NP * 3;
    float px = base[i * 3 + 0], py = base[i * 3 + 1], pz = base[i * 3 + 2];
    float bd[KNN];
    int   bi[KNN];
#pragma unroll
    for (int j = 0; j < KNN; ++j) { bd[j] = 3.4e38f; bi[j] = 0; }

    for (int tile = 0; tile < NP; tile += 256) {
        __syncthreads();
        int c = tile + threadIdx.x;
        sx[threadIdx.x] = base[c * 3 + 0];
        sy[threadIdx.x] = base[c * 3 + 1];
        sz[threadIdx.x] = base[c * 3 + 2];
        __syncthreads();
        for (int t = 0; t < 256; ++t) {
            float dx = px - sx[t], dy = py - sy[t], dz = pz - sz[t];
            float d = dx * dx + dy * dy + dz * dz;
            if (d < bd[KNN - 1]) {
                float cd = d; int ci = tile + t;
#pragma unroll
                for (int j = 0; j < KNN; ++j) {
                    if (cd < bd[j]) {
                        float tf = bd[j]; bd[j] = cd; cd = tf;
                        int   ti = bi[j]; bi[j] = ci; ci = ti;
                    }
                }
            }
        }
    }
    int* orow = idx + ((size_t)b * NP + i) * KNN;
#pragma unroll
    for (int j = 0; j < KNN; ++j) orow[j] = bi[j];
}

// ---------------- kernel 2: x = feat@fc1+b ; q/k/v = x@W -------------------
// 256 threads = 8 waves; each wave owns 16 consecutive rows of [B*N].
__global__ void proj_kernel(const float* __restrict__ feat,
                            const float* __restrict__ fc1_b,
                            const bf16* __restrict__ fc1T,
                            const bf16* __restrict__ wqT,
                            const bf16* __restrict__ wkT,
                            const bf16* __restrict__ wvT,
                            float* __restrict__ qout,
                            float* __restrict__ kout,
                            float* __restrict__ vout) {
    __shared__ __align__(16) bf16 xs[8][16 * DM];   // 32 KB
    int wave = threadIdx.x >> 5, lane = threadIdx.x & 31;
    int hl = lane >> 4, n = lane & 15;
    size_t row0 = ((size_t)blockIdx.x * 8 + wave) * 16;

    // A fragments from features [16 x 64] f32 (K = 64 -> 2 chunks)
    v16bf afr[2];
    const float* arow = feat + (row0 + n) * DP;
#pragma unroll
    for (int kk = 0; kk < 2; ++kk) {
        const float* p0 = arow + kk * 32 + hl * 8;
        const float* p1 = arow + kk * 32 + 16 + hl * 8;
#pragma unroll
        for (int e = 0; e < 8; ++e) {
            afr[kk][e]     = (bf16)p0[e];
            afr[kk][8 + e] = (bf16)p1[e];
        }
    }

    bf16* myxs = xs[wave];
    // x = feat @ fc1 + fc1_b  -> staged bf16 in LDS (row-major [16][128])
#pragma unroll
    for (int t = 0; t < 8; ++t) {
        int col = t * 16 + n;
        float bb = fc1_b[col];
        v8f acc;
#pragma unroll
        for (int r = 0; r < 8; ++r) acc[r] = bb;
#pragma unroll
        for (int kk = 0; kk < 2; ++kk) {
            v16bf bfr = load_bfrag_g(fc1T, col, kk, hl, DP);
            acc = __builtin_amdgcn_wmma_f32_16x16x32_bf16(
                false, afr[kk], false, bfr, (short)0, acc, false, false);
        }
#pragma unroll
        for (int r = 0; r < 8; ++r)
            myxs[(r + 8 * hl) * DM + col] = (bf16)acc[r];
    }
    __syncthreads();

    // reload x as A fragments (K = 128 -> 4 chunks)
    v16bf xfr[4];
#pragma unroll
    for (int kk = 0; kk < 4; ++kk) xfr[kk] = load_afrag_lds(myxs, n, kk, hl);

    const bf16* Ws[3] = { wqT, wkT, wvT };
    float*      Os[3] = { qout, kout, vout };
#pragma unroll
    for (int m = 0; m < 3; ++m) {
        const bf16* W = Ws[m];
        float*      O = Os[m];
#pragma unroll
        for (int t = 0; t < 8; ++t) {
            int col = t * 16 + n;
            v8f acc;
#pragma unroll
            for (int r = 0; r < 8; ++r) acc[r] = 0.0f;
#pragma unroll
            for (int kk = 0; kk < 4; ++kk) {
                v16bf bfr = load_bfrag_g(W, col, kk, hl, DM);
                acc = __builtin_amdgcn_wmma_f32_16x16x32_bf16(
                    false, xfr[kk], false, bfr, (short)0, acc, false, false);
            }
#pragma unroll
            for (int r = 0; r < 8; ++r)
                O[(row0 + r + 8 * hl) * DM + col] = acc[r];
        }
    }
}

// ---------------- kernel 3: fused pos_enc / gamma / softmax / output -------
// 128 threads = 4 waves; each wave owns one query point.
// Dynamic LDS layout (bytes):
//   [0)       d2T|g1T|g2T bf16 copies   3*32768 = 98304   (one TDM load)
//   [98304)   d1s  f32 3*128            1536
//   [99840)   rS   f32 4*128            2048
//   [101888)  hS   bf16 4*16*128        16384
//   [118272)  tS   bf16 4*16*128        16384
//   [134656)  pS   bf16 4*16*128        16384
//   total = 151040 bytes
#define SM_TOTAL 151040

__global__ void fused_attn_kernel(const float* __restrict__ xyz,
                                  const float* __restrict__ feat,
                                  const float* __restrict__ d1_w,
                                  const float* __restrict__ d1_b,
                                  const float* __restrict__ d2_b,
                                  const float* __restrict__ g1_b,
                                  const float* __restrict__ g2_b,
                                  const float* __restrict__ fc2_w,
                                  const float* __restrict__ fc2_b,
                                  const bf16* __restrict__ wsrc,  // d2T|g1T|g2T
                                  const int* __restrict__ idxbuf,
                                  const float* __restrict__ qb,
                                  const float* __restrict__ kb,
                                  const float* __restrict__ vb,
                                  float* __restrict__ out,
                                  float* __restrict__ attn_out) {
    extern __shared__ __align__(16) char smem[];
    bf16*  wD2 = (bf16*)smem;
    bf16*  wG1 = wD2 + 16384;
    bf16*  wG2 = wG1 + 16384;
    float* d1s = (float*)(smem + 98304);
    float* rS  = (float*)(smem + 99840);
    bf16*  hS  = (bf16*)(smem + 101888);
    bf16*  tS  = (bf16*)(smem + 118272);
    bf16*  pS  = (bf16*)(smem + 134656);

    int wave = threadIdx.x >> 5, lane = threadIdx.x & 31;
    int hl = lane >> 4, n = lane & 15;

    // ---- stage the 96KB of gamma/pos weights into LDS ----
#if USE_TDM
    if (wave == 0) {
        tdm_load_1d_bf16(wsrc, (unsigned)(uintptr_t)smem, 3 * 16384);
    }
#else
    {
        const float4* gsrc4 = (const float4*)wsrc;
        float4* dst4 = (float4*)smem;
        for (int o = threadIdx.x; o < (3 * 32768) / 16; o += blockDim.x)
            dst4[o] = gsrc4[o];
    }
#endif
    for (int o = threadIdx.x; o < 3 * DM; o += blockDim.x) d1s[o] = d1_w[o];

    // ---- overlap: per-point indices / relative coords while TDM runs ----
    size_t pt = (size_t)blockIdx.x * 4 + wave;   // flat point id in [0, B*N)
    int b = (int)(pt >> 12);
    int i = (int)(pt & (NP - 1));

    const int* irow = idxbuf + pt * KNN;
    int idm = irow[n];                   // neighbor row for A-layout (m = n)
    int idr[8];
#pragma unroll
    for (int r = 0; r < 8; ++r) idr[r] = irow[r + 8 * hl];

    const float* xb = xyz + (size_t)b * NP * 3;
    float cx = xb[i * 3 + 0], cy = xb[i * 3 + 1], cz = xb[i * 3 + 2];
    float rx = cx - xb[idm * 3 + 0];
    float ry = cy - xb[idm * 3 + 1];
    float rz = cz - xb[idm * 3 + 2];

#if USE_TDM
    if (wave == 0) __builtin_amdgcn_s_wait_tensorcnt(0);
#endif
    __syncthreads();   // weights + d1s visible to all waves

    // pos1 = relu(rel @ d1 + d1_b), generated directly in A-fragment layout
    v16bf pA[4];
#pragma unroll
    for (int kk = 0; kk < 4; ++kk) {
#pragma unroll
        for (int e = 0; e < 16; ++e) {
            int k = kk * 32 + ((e < 8) ? (8 * hl + e) : (8 + 8 * hl + e));
            float v = d1_b[k] + rx * d1s[k] + ry * d1s[DM + k] + rz * d1s[2 * DM + k];
            pA[kk][e] = (bf16)fmaxf(v, 0.0f);
        }
    }

    bf16* hL = hS + wave * 16 * DM;
    bf16* tL = tS + wave * 16 * DM;
    bf16* pL = pS + wave * 16 * DM;
    const float* qrow  = qb + pt * DM;
    const float* kbase = kb + (size_t)b * NP * DM;
    const float* vbase = vb + (size_t)b * NP * DM;

    // pos_enc = pos1 @ d2 + d2_b ; h = q - k_gather + pos_enc  (C/D layout)
#pragma unroll
    for (int t = 0; t < 8; ++t) {
        int col = t * 16 + n;
        float bb = d2_b[col];
        v8f acc;
#pragma unroll
        for (int r = 0; r < 8; ++r) acc[r] = bb;
#pragma unroll
        for (int kk = 0; kk < 4; ++kk) {
            v16bf bfr = load_bfrag_l(wD2, col, kk, hl);
            acc = __builtin_amdgcn_wmma_f32_16x16x32_bf16(
                false, pA[kk], false, bfr, (short)0, acc, false, false);
        }
        float qv = qrow[col];
#pragma unroll
        for (int r = 0; r < 8; ++r) {
            float pe = acc[r];
            int row = r + 8 * hl;
            pL[row * DM + col] = (bf16)pe;
            float kf = kbase[(size_t)idr[r] * DM + col];
            hL[row * DM + col] = (bf16)(qv - kf + pe);
        }
    }
    __syncthreads();

    // t1 = relu(h @ g1 + g1_b)
    v16bf hA[4];
#pragma unroll
    for (int kk = 0; kk < 4; ++kk) hA[kk] = load_afrag_lds(hL, n, kk, hl);
#pragma unroll
    for (int t = 0; t < 8; ++t) {
        int col = t * 16 + n;
        float bb = g1_b[col];
        v8f acc;
#pragma unroll
        for (int r = 0; r < 8; ++r) acc[r] = bb;
#pragma unroll
        for (int kk = 0; kk < 4; ++kk) {
            v16bf bfr = load_bfrag_l(wG1, col, kk, hl);
            acc = __builtin_amdgcn_wmma_f32_16x16x32_bf16(
                false, hA[kk], false, bfr, (short)0, acc, false, false);
        }
#pragma unroll
        for (int r = 0; r < 8; ++r)
            tL[(r + 8 * hl) * DM + col] = (bf16)fmaxf(acc[r], 0.0f);
    }
    __syncthreads();

    // attn_pre = t1 @ g2 + g2_b ; softmax over K (rows) ; weighted sum
    v16bf tA[4];
#pragma unroll
    for (int kk = 0; kk < 4; ++kk) tA[kk] = load_afrag_lds(tL, n, kk, hl);

    float* attnrow = attn_out + pt * KNN * DM;
    float res[8];
#pragma unroll
    for (int t = 0; t < 8; ++t) {
        int col = t * 16 + n;
        float bb = g2_b[col];
        v8f acc;
#pragma unroll
        for (int r = 0; r < 8; ++r) acc[r] = bb;
#pragma unroll
        for (int kk = 0; kk < 4; ++kk) {
            v16bf bfr = load_bfrag_l(wG2, col, kk, hl);
            acc = __builtin_amdgcn_wmma_f32_16x16x32_bf16(
                false, tA[kk], false, bfr, (short)0, acc, false, false);
        }
        // softmax over 16 rows of this column (half in lane, half in lane^16)
        float sv[8], m8 = -3.4e38f;
#pragma unroll
        for (int r = 0; r < 8; ++r) { sv[r] = acc[r] * SCALE; m8 = fmaxf(m8, sv[r]); }
        float mall = fmaxf(m8, __shfl_xor(m8, 16, 32));
        float s8 = 0.0f;
#pragma unroll
        for (int r = 0; r < 8; ++r) { sv[r] = __expf(sv[r] - mall); s8 += sv[r]; }
        float inv = 1.0f / (s8 + __shfl_xor(s8, 16, 32));
        float rsum = 0.0f;
#pragma unroll
        for (int r = 0; r < 8; ++r) {
            int row = r + 8 * hl;
            float a = sv[r] * inv;
            attnrow[row * DM + col] = a;
            float pe = (float)pL[row * DM + col];
            float vv = vbase[(size_t)idr[r] * DM + col];
            rsum += a * (vv + pe);
        }
        res[t] = rsum + __shfl_xor(rsum, 16, 32);
    }

    if (hl == 0) {
#pragma unroll
        for (int t = 0; t < 8; ++t) rS[wave * DM + t * 16 + n] = res[t];
    }
    __syncthreads();

    // out = res @ fc2_w + fc2_b + features   (64 outputs, 2 per lane)
    const float* prow = feat + pt * DP;
    float* orow = out + pt * DP;
#pragma unroll
    for (int oo = 0; oo < 2; ++oo) {
        int o = lane * 2 + oo;
        float s = fc2_b[o] + prow[o];
        for (int c = 0; c < DM; ++c) s += rS[wave * DM + c] * fc2_w[c * DP + o];
        orow[o] = s;
    }
}

// ---------------------------------------------------------------------------
extern "C" void kernel_launch(void* const* d_in, const int* in_sizes, int n_in,
                              void* d_out, int out_size, void* d_ws, size_t ws_size,
                              hipStream_t stream) {
    const float* xyz   = (const float*)d_in[0];
    const float* feat  = (const float*)d_in[1];
    const float* fc1_w = (const float*)d_in[2];
    const float* fc1_b = (const float*)d_in[3];
    const float* fc2_w = (const float*)d_in[4];
    const float* fc2_b = (const float*)d_in[5];
    const float* d1_w  = (const float*)d_in[6];
    const float* d1_b  = (const float*)d_in[7];
    const float* d2_w  = (const float*)d_in[8];
    const float* d2_b  = (const float*)d_in[9];
    const float* g1_w  = (const float*)d_in[10];
    const float* g1_b  = (const float*)d_in[11];
    const float* g2_w  = (const float*)d_in[12];
    const float* g2_b  = (const float*)d_in[13];
    const float* wq_w  = (const float*)d_in[14];
    const float* wk_w  = (const float*)d_in[15];
    const float* wv_w  = (const float*)d_in[16];

    char* ws = (char*)d_ws;
    const size_t MB = 1u << 20;
    int*   idx = (int*)(ws + 0);                         // 1 MB
    float* q   = (float*)(ws + 1 * MB);                  // 8 MB
    float* kp  = (float*)(ws + 9 * MB);                  // 8 MB
    float* vp  = (float*)(ws + 17 * MB);                 // 8 MB
    bf16*  wbf = (bf16*)(ws + 25 * MB);                  // bf16 weights
    bf16* fc1T = wbf;                  // [128][64]
    bf16* wqT  = fc1T + DM * DP;       // [128][128]
    bf16* wkT  = wqT  + DM * DM;
    bf16* wvT  = wkT  + DM * DM;
    bf16* d2T  = wvT  + DM * DM;       // d2T|g1T|g2T contiguous (one TDM load)
    bf16* g1T  = d2T  + DM * DM;
    bf16* g2T  = g1T  + DM * DM;

    // 0) weight transpose + bf16 conversion
    transpose_bf16_kernel<<<(DP * DM + 255) / 256, 256, 0, stream>>>(fc1_w, fc1T, DP, DM);
    transpose_bf16_kernel<<<(DM * DM + 255) / 256, 256, 0, stream>>>(wq_w, wqT, DM, DM);
    transpose_bf16_kernel<<<(DM * DM + 255) / 256, 256, 0, stream>>>(wk_w, wkT, DM, DM);
    transpose_bf16_kernel<<<(DM * DM + 255) / 256, 256, 0, stream>>>(wv_w, wvT, DM, DM);
    transpose_bf16_kernel<<<(DM * DM + 255) / 256, 256, 0, stream>>>(d2_w, d2T, DM, DM);
    transpose_bf16_kernel<<<(DM * DM + 255) / 256, 256, 0, stream>>>(g1_w, g1T, DM, DM);
    transpose_bf16_kernel<<<(DM * DM + 255) / 256, 256, 0, stream>>>(g2_w, g2T, DM, DM);

    // 1) KNN
    knn_kernel<<<dim3(NP / 256, BN), 256, 0, stream>>>(xyz, idx);

    // 2) projections (B*N rows / 16 per wave / 8 waves per block)
    proj_kernel<<<(BN * NP) / (16 * 8), 256, 0, stream>>>(
        feat, fc1_b, fc1T, wqT, wkT, wvT, q, kp, vp);

    // 3) fused attention (1 wave per point, 4 waves per block, 147.5KB LDS)
    float* out_p  = (float*)d_out;
    float* attn_p = out_p + (size_t)BN * NP * DP;
    fused_attn_kernel<<<(BN * NP) / 4, 128, SM_TOTAL, stream>>>(
        xyz, feat, d1_w, d1_b, d2_b, g1_b, g2_b, fc2_w, fc2_b,
        d2T, idx, q, kp, vp, out_p, attn_p);
}
